// Head_11364483465679
// MI455X (gfx1250) — matrix operand order repaired and checked by reading, not compiled, not run on hardware
//
#include <hip/hip_runtime.h>

typedef __attribute__((ext_vector_type(16))) _Float16 v16h;
typedef __attribute__((ext_vector_type(8)))  _Float16 v8h;
typedef __attribute__((ext_vector_type(8)))  float    v8f;

static constexpr int NE = 1024;   // n_embed
static constexpr int HS = 64;     // head size
static constexpr int TT = 2048;   // sequence length
static constexpr int NB = 8;      // batch

#define WMMA_F32_F16(a, b, c) \
  __builtin_amdgcn_wmma_f32_16x16x32_f16(false, (a), false, (b), (short)0, (c), false, false)

// ---- fragment loaders (contiguous k order, same in A and B) ----

// A-matrix 16x32 f16: lane l holds row (l&15); K = klo..klo+15 contiguous,
// klo = (l>>4)*16.
__device__ __forceinline__ v16h load_a_f16(const _Float16* base, int ld) {
  const int l   = threadIdx.x & 31;
  const int r   = l & 15;
  const int klo = (l >> 4) * 16;
  const _Float16* p = base + (size_t)r * ld + klo;
  v8h lo = *(const v8h*)(p);
  v8h hi = *(const v8h*)(p + 8);
  return __builtin_shufflevector(lo, hi, 0,1,2,3,4,5,6,7,8,9,10,11,12,13,14,15);
}

// Same A layout but source is fp32 (converted to f16 on load).
__device__ __forceinline__ v16h load_a_f32cvt(const float* base, int ld) {
  const int l   = threadIdx.x & 31;
  const int r   = l & 15;
  const int klo = (l >> 4) * 16;
  const float* p = base + (size_t)r * ld + klo;
  v16h out;
#pragma unroll
  for (int i = 0; i < 8; ++i) {
    out[i]     = (_Float16)p[i];
    out[i + 8] = (_Float16)p[i + 8];
  }
  return out;
}

// B-matrix 32x16 f16: lane l holds column N=(l&15); K = (l>>4)*16 .. +15
// contiguous (2 per VGPR).  base points at B^T storage: row = N, stride ld.
__device__ __forceinline__ v16h load_b_f16(const _Float16* base, int ld) {
  const int l = threadIdx.x & 31;
  const _Float16* p = base + (size_t)(l & 15) * ld + ((l >> 4) * 16);
  return *(const v16h*)p;
}

// ---- kernel 1: W[1024,64] fp32 -> WT[64,1024] f16 (x3 projections) ----
__global__ void wtrans_kernel(const float* __restrict__ Wq,
                              const float* __restrict__ Wk,
                              const float* __restrict__ Wv,
                              _Float16* __restrict__ WT) {
  int idx = blockIdx.x * blockDim.x + threadIdx.x;
  if (idx >= 3 * HS * NE) return;
  int p   = idx / (HS * NE);
  int rem = idx % (HS * NE);
  int n   = rem / NE;   // head col
  int kk  = rem % NE;   // embed index
  const float* W = (p == 0) ? Wq : ((p == 1) ? Wk : Wv);
  WT[idx] = (_Float16)W[(size_t)kk * HS + n];
}

// ---- kernel 2: projections.  grid = (B*T/16, 3), block = 32 (one wave) ----
// q,k stored row-major f16 [B*T,64]; v stored transposed f16 [B][64][T].
__global__ void proj_kernel(const float* __restrict__ x,
                            const _Float16* __restrict__ WT,
                            _Float16* __restrict__ q,
                            _Float16* __restrict__ k,
                            _Float16* __restrict__ vT) {
  const int tile = blockIdx.x;       // 16-row tile over B*T
  const int pj   = blockIdx.y;       // 0=q 1=k 2=v
  const int l    = threadIdx.x & 31;
  const int n    = l & 15;
  const int mhi  = (l >> 4) * 8;
  const size_t row0 = (size_t)tile * 16;
  const _Float16* Wp = WT + (size_t)pj * HS * NE;
  const float* xb = x + row0 * NE;

  v8f acc[4] = {};
  for (int k0 = 0; k0 < NE; k0 += 32) {
    v16h a = load_a_f32cvt(xb + k0, NE);
#pragma unroll
    for (int j = 0; j < 4; ++j) {
      v16h bf = load_b_f16(Wp + (size_t)(j * 16) * NE + k0, NE);
      acc[j] = WMMA_F32_F16(a, bf, acc[j]);
    }
  }

  if (pj < 2) {
    _Float16* dst = (pj == 0) ? q : k;
#pragma unroll
    for (int j = 0; j < 4; ++j)
#pragma unroll
      for (int r = 0; r < 8; ++r)
        dst[(row0 + mhi + r) * HS + j * 16 + n] = (_Float16)acc[j][r];
  } else {
    const int b  = tile / (TT / 16);
    const int t0 = (tile % (TT / 16)) * 16;
#pragma unroll
    for (int j = 0; j < 4; ++j) {
      v8h pk;
#pragma unroll
      for (int r = 0; r < 8; ++r) pk[r] = (_Float16)acc[j][r];
      *(v8h*)(vT + ((size_t)b * HS + j * 16 + n) * TT + t0 + mhi) = pk;
    }
  }
}

// ---- kernel 3: causal flash attention. grid = (T/16, B), block = 32 ----
__global__ void flash_kernel(const _Float16* __restrict__ q,
                             const _Float16* __restrict__ k,
                             const _Float16* __restrict__ vT,
                             float* __restrict__ out) {
  __shared__ alignas(32) _Float16 Plds[16 * 64];
  const int l   = threadIdx.x & 31;
  const int n   = l & 15;
  const int mhi = (l >> 4) * 8;
  const int q0  = blockIdx.x * 16;
  const int b   = blockIdx.y;

  const _Float16* qb = q  + ((size_t)b * TT + q0) * HS;
  const _Float16* kb = k  + (size_t)b * TT * HS;
  const _Float16* vb = vT + (size_t)b * HS * TT;

  // q A-fragments (head 0..31 / 32..63), hoisted for the whole key loop
  const v16h aq0 = load_a_f16(qb, HS);
  const v16h aq1 = load_a_f16(qb + 32, HS);

  float mrun[8], lrun[8];
  v8f O[4] = {};
#pragma unroll
  for (int r = 0; r < 8; ++r) { mrun[r] = -1e30f; lrun[r] = 0.f; }

  const float scale = 0.03125f;   // 1024^-0.5 (reference scales by C, not H)
  const int nkb = q0 / 64 + 1;    // causal: only key blocks up to the diagonal

  for (int kbi = 0; kbi < nkb; ++kbi) {
    const int s0 = kbi * 64;
    float Sv[4][8];

    // S = q @ k^T for a 16x64 block (4 n-subtiles, K=64 head dim = 2 WMMAs)
#pragma unroll
    for (int js = 0; js < 4; ++js) {
      const _Float16* krow = kb + (size_t)(s0 + js * 16) * HS;
      v16h bk0 = load_b_f16(krow, HS);
      v16h bk1 = load_b_f16(krow + 32, HS);
      v8f s = {};
      s = WMMA_F32_F16(aq0, bk0, s);
      s = WMMA_F32_F16(aq1, bk1, s);
#pragma unroll
      for (int r = 0; r < 8; ++r) {
        const int kj = s0 + js * 16 + n;
        const int qi = q0 + mhi + r;
        Sv[js][r] = (kj > qi) ? -1e30f : s[r] * scale;
      }
    }

    // per-row block max (rows live across a 16-lane group)
    float alpha[8];
#pragma unroll
    for (int r = 0; r < 8; ++r) {
      float m = fmaxf(fmaxf(Sv[0][r], Sv[1][r]), fmaxf(Sv[2][r], Sv[3][r]));
      for (int off = 1; off < 16; off <<= 1)
        m = fmaxf(m, __shfl_xor(m, off, 32));
      float mnew = fmaxf(mrun[r], m);
      alpha[r]   = __expf(mrun[r] - mnew);
      mrun[r]    = mnew;
    }

    // P = exp(S - m), write C-layout -> LDS (row-major f16), row sums
#pragma unroll
    for (int r = 0; r < 8; ++r) {
      float su = 0.f;
#pragma unroll
      for (int js = 0; js < 4; ++js) {
        float pv = __expf(Sv[js][r] - mrun[r]);
        su += pv;
        Plds[(mhi + r) * 64 + js * 16 + n] = (_Float16)pv;
      }
      for (int off = 1; off < 16; off <<= 1)
        su += __shfl_xor(su, off, 32);
      lrun[r] = lrun[r] * alpha[r] + su;
    }

    // rescale running output
#pragma unroll
    for (int j = 0; j < 4; ++j)
#pragma unroll
      for (int r = 0; r < 8; ++r)
        O[j][r] *= alpha[r];

    asm volatile("s_wait_dscnt 0" ::: "memory");  // LDS stores visible to loads

    // O += P @ V : A from LDS (16x64 keys -> two 16x32 frags), B from vT
    v16h ap0 = load_a_f16(Plds, 64);
    v16h ap1 = load_a_f16(Plds + 32, 64);
#pragma unroll
    for (int j = 0; j < 4; ++j) {
      const _Float16* vrow = vb + (size_t)(j * 16) * TT + s0;
      v16h bv0 = load_b_f16(vrow, TT);
      v16h bv1 = load_b_f16(vrow + 32, TT);
      O[j] = WMMA_F32_F16(ap0, bv0, O[j]);
      O[j] = WMMA_F32_F16(ap1, bv1, O[j]);
    }
  }

  // out = O / l  (fp32, row-major [B,T,64])
  float* ob = out + ((size_t)b * TT + q0) * HS;
#pragma unroll
  for (int j = 0; j < 4; ++j)
#pragma unroll
    for (int r = 0; r < 8; ++r)
      ob[(size_t)(mhi + r) * HS + j * 16 + n] = O[j][r] / lrun[r];
}

extern "C" void kernel_launch(void* const* d_in, const int* in_sizes, int n_in,
                              void* d_out, int out_size, void* d_ws, size_t ws_size,
                              hipStream_t stream) {
  (void)in_sizes; (void)n_in; (void)out_size; (void)ws_size;
  const float* x  = (const float*)d_in[0];
  const float* Wq = (const float*)d_in[1];
  const float* Wk = (const float*)d_in[2];
  const float* Wv = (const float*)d_in[3];
  float* out = (float*)d_out;

  char* ws = (char*)d_ws;
  size_t off = 0;
  _Float16* WT  = (_Float16*)(ws + off);
  off += ((size_t)3 * HS * NE * sizeof(_Float16) + 255) & ~(size_t)255;
  _Float16* qws = (_Float16*)(ws + off); off += (size_t)NB * TT * HS * sizeof(_Float16);
  _Float16* kws = (_Float16*)(ws + off); off += (size_t)NB * TT * HS * sizeof(_Float16);
  _Float16* vTs = (_Float16*)(ws + off);

  const int nW = 3 * HS * NE;
  wtrans_kernel<<<(nW + 255) / 256, 256, 0, stream>>>(Wq, Wk, Wv, WT);

  dim3 pg(NB * TT / 16, 3);
  proj_kernel<<<pg, 32, 0, stream>>>(x, WT, qws, kws, vTs);

  dim3 fg(TT / 16, NB);
  flash_kernel<<<fg, 32, 0, stream>>>(qws, kws, vTs, out);
}
